// EncoderLayer_43722767073434
// MI455X (gfx1250) — matrix operand order repaired and checked by reading, not compiled, not run on hardware
//
#include <hip/hip_runtime.h>
#include <stdint.h>

// EncoderLayer, algebraically restructured (exact math):
//   scores[h] = (q G[h] k^T + u ⊕ t + c)/8,  G[h]=Wq[h]Wk[h]^T (64x64)
//   pooled    = Σ_h ( (w_h · v) Wv[h] + (Σ w_h) bv[h] ),  w_h[m]=Σ_n attn[h,n,m]
// Dominant kernel k3 uses v_wmma_f32_16x16x32_bf16, two independent m-tiles
// in flight per iteration to hide WMMA hazard latency.

#define HH 16
#define DKv 64
#define Dm 1024
#define NN 2048
#define MM 2048
#define SROW 20   // floats per m-row in LDS score buffer (16 + 4 pad; gcd(20,64)=4 -> 16 banks)

typedef __attribute__((ext_vector_type(16))) __bf16 v16bf;
typedef __attribute__((ext_vector_type(8)))  __bf16 v8bf;
typedef __attribute__((ext_vector_type(8)))  float  v8f;

union AV { v16bf v; v8bf h[2]; };

// ---------------- K1: per-head G, a, b, c; zero w/pooled ----------------
__global__ __launch_bounds__(256) void k1_prep(
    const float* __restrict__ Wq, const float* __restrict__ bq,
    const float* __restrict__ Wk, const float* __restrict__ bk,
    float* __restrict__ G, float* __restrict__ av, float* __restrict__ bvw,
    float* __restrict__ cb, float* __restrict__ w, float* __restrict__ pooled)
{
  const int h = blockIdx.x, tid = threadIdx.x;
  const float* Wqh = Wq + (size_t)h * DKv * Dm;
  const float* Wkh = Wk + (size_t)h * DKv * Dm;
  for (int e = tid; e < DKv * DKv; e += 256) {
    const int i = e >> 6, j = e & 63;
    const float* wqi = Wqh + (size_t)i * Dm;
    const float* wkj = Wkh + (size_t)j * Dm;
    float acc = 0.f;
    for (int d = 0; d < Dm; ++d) acc += wqi[d] * wkj[d];
    G[(size_t)h * DKv * DKv + e] = acc;
  }
  if (tid < 64) {
    const float* wqi = Wqh + (size_t)tid * Dm;
    float acc = 0.f;
    for (int d = 0; d < Dm; ++d) acc += wqi[d] * bk[h * Dm + d];
    av[h * 64 + tid] = acc;
  } else if (tid < 128) {
    const int j = tid - 64;
    const float* wkj = Wkh + (size_t)j * Dm;
    float acc = 0.f;
    for (int d = 0; d < Dm; ++d) acc += wkj[d] * bq[h * Dm + d];
    bvw[h * 64 + j] = acc;
  } else if (tid == 128) {
    float acc = 0.f;
    for (int d = 0; d < Dm; ++d) acc += bq[h * Dm + d] * bk[h * Dm + d];
    cb[h] = acc;
  }
  for (int m = tid; m < MM; m += 256) w[h * MM + m] = 0.f;
  if (h == 0) for (int d = tid; d < Dm; d += 256) pooled[d] = 0.f;
}

// ---------------- K2: qG=q@G[h] (bf16), u, t, k->bf16 ----------------
__global__ __launch_bounds__(256) void k2_qg(
    const float* __restrict__ q, const float* __restrict__ k,
    const float* __restrict__ G, const float* __restrict__ av,
    const float* __restrict__ bvw,
    __bf16* __restrict__ qG, __bf16* __restrict__ kbf,
    float* __restrict__ u, float* __restrict__ t)
{
  __shared__ float qs[64][65], ks[64][65], Gs[64][65];
  __shared__ float as_[64], bs_[64];
  const int tid = threadIdx.x;
  const int n0 = blockIdx.x * 64, h = blockIdx.y;
  for (int e = tid; e < 4096; e += 256) {
    const int r = e >> 6, c = e & 63;
    qs[r][c] = q[(size_t)(n0 + r) * 64 + c];
    ks[r][c] = k[(size_t)(n0 + r) * 64 + c];
    Gs[r][c] = G[(size_t)h * 4096 + e];
  }
  if (tid < 64) { as_[tid] = av[h * 64 + tid]; bs_[tid] = bvw[h * 64 + tid]; }
  __syncthreads();
  const int nl = tid >> 2, qd = tid & 3, i0 = qd * 16;
  for (int i = i0; i < i0 + 16; ++i) {
    float acc = 0.f;
    for (int j = 0; j < 64; ++j) acc += qs[nl][j] * Gs[j][i];
    qG[((size_t)h * NN + n0 + nl) * 64 + i] = (__bf16)acc;
  }
  if (qd == 0) {
    float acc = 0.f;
    for (int j = 0; j < 64; ++j) acc += qs[nl][j] * as_[j];
    u[h * NN + n0 + nl] = acc;
  } else if (qd == 1) {
    float acc = 0.f;
    for (int j = 0; j < 64; ++j) acc += ks[nl][j] * bs_[j];
    t[h * MM + n0 + nl] = acc;
  } else if (qd == 2 && h == 0) {
    for (int i = 0; i < 64; ++i) kbf[(size_t)(n0 + nl) * 64 + i] = (__bf16)ks[nl][i];
  }
}

// ---------------- K3: WMMA scores + softmax + w accumulation ----------------
// LDS score layout is m-major: sS[m*SROW + n], n in [0,16). Each lane's 8
// accumulator values are contiguous -> 2x ds_store_b128 per tile, bank-clean.
__global__ __launch_bounds__(256) void k3_attn(
    const __bf16* __restrict__ qG, const __bf16* __restrict__ kbf,
    const float* __restrict__ u, const float* __restrict__ t,
    const float* __restrict__ cb, float* __restrict__ w)
{
  extern __shared__ float sm[];
  float* sS   = sm;                    // MM * SROW scores (m-major)
  float* red  = sS + MM * SROW;        // 16x16 reduction partials
  float* rowM = red + 256;             // 16
  float* rowR = rowM + 16;             // 16
  const int tid = threadIdx.x;
  const int h = blockIdx.y, nb = blockIdx.x;
  const int lane = tid & 31, wid = tid >> 5;
  const int hi = lane >> 4, lr = lane & 15;
  const float cbh = cb[h];

  // Per-lane bias values for the 8 output rows this lane owns (fold c[h] in).
  float u8v[8];
  {
    const float* ub = u + h * NN + nb * 16 + 8 * hi;
#pragma unroll
    for (int r = 0; r < 8; ++r) u8v[r] = ub[r] + cbh;
  }

  // A fragment: 16x64 bf16 block of qG rows (16 contiguous K per lane-half
  // per K-32 slab, matching B).
  const __bf16* qrow = qG + ((size_t)h * NN + nb * 16 + lr) * 64;
  AV aLo, aHi;
  aLo.h[0] = *(const v8bf*)(qrow + hi * 16);
  aLo.h[1] = *(const v8bf*)(qrow + 8 + hi * 16);
  aHi.h[0] = *(const v8bf*)(qrow + 32 + hi * 16);
  aHi.h[1] = *(const v8bf*)(qrow + 40 + hi * 16);

  // Two independent m-tiles per iteration: c0/c1 chains hide WMMA hazards.
  for (int tile = wid; tile < 128; tile += 16) {
    const int m0a = tile * 16, m0b = (tile + 8) * 16;
    const __bf16* ka = kbf + (size_t)(m0a + lr) * 64;
    const __bf16* kb = kbf + (size_t)(m0b + lr) * 64;
    if (tile + 16 < 128) {
      __builtin_prefetch(kbf + (size_t)((tile + 16) * 16 + lr) * 64, 0, 1);
      __builtin_prefetch(kbf + (size_t)((tile + 24) * 16 + lr) * 64, 0, 1);
    }
    // B fragment: lane = key column, 16 contiguous K features per lane-half.
    v16bf b0lo = *(const v16bf*)(ka + hi * 16);
    v16bf b0hi = *(const v16bf*)(ka + 32 + hi * 16);
    v16bf b1lo = *(const v16bf*)(kb + hi * 16);
    v16bf b1hi = *(const v16bf*)(kb + 32 + hi * 16);
    v8f c0 = {}, c1 = {};
    c0 = __builtin_amdgcn_wmma_f32_16x16x32_bf16(false, aLo.v, false, b0lo,
                                                 (short)0, c0, false, false);
    c1 = __builtin_amdgcn_wmma_f32_16x16x32_bf16(false, aLo.v, false, b1lo,
                                                 (short)0, c1, false, false);
    c0 = __builtin_amdgcn_wmma_f32_16x16x32_bf16(false, aHi.v, false, b0hi,
                                                 (short)0, c0, false, false);
    c1 = __builtin_amdgcn_wmma_f32_16x16x32_bf16(false, aHi.v, false, b1hi,
                                                 (short)0, c1, false, false);
    const float ta = t[h * MM + m0a + lr];
    const float tb = t[h * MM + m0b + lr];
    {
      float4 lo = { (c0[0] + u8v[0] + ta) * 0.125f, (c0[1] + u8v[1] + ta) * 0.125f,
                    (c0[2] + u8v[2] + ta) * 0.125f, (c0[3] + u8v[3] + ta) * 0.125f };
      float4 hi4 = { (c0[4] + u8v[4] + ta) * 0.125f, (c0[5] + u8v[5] + ta) * 0.125f,
                     (c0[6] + u8v[6] + ta) * 0.125f, (c0[7] + u8v[7] + ta) * 0.125f };
      float* dst = &sS[(m0a + lr) * SROW + 8 * hi];
      *(float4*)dst = lo;
      *(float4*)(dst + 4) = hi4;
    }
    {
      float4 lo = { (c1[0] + u8v[0] + tb) * 0.125f, (c1[1] + u8v[1] + tb) * 0.125f,
                    (c1[2] + u8v[2] + tb) * 0.125f, (c1[3] + u8v[3] + tb) * 0.125f };
      float4 hi4 = { (c1[4] + u8v[4] + tb) * 0.125f, (c1[5] + u8v[5] + tb) * 0.125f,
                     (c1[6] + u8v[6] + tb) * 0.125f, (c1[7] + u8v[7] + tb) * 0.125f };
      float* dst = &sS[(m0b + lr) * SROW + 8 * hi];
      *(float4*)dst = lo;
      *(float4*)(dst + 4) = hi4;
    }
  }
  __syncthreads();

  // softmax over m per row n (exact, two reduction passes in LDS)
  const int row = tid & 15, chk = tid >> 4;
  float mx = -3.4e38f;
  for (int m = chk * 128; m < chk * 128 + 128; ++m)
    mx = fmaxf(mx, sS[m * SROW + row]);
  red[chk * 16 + row] = mx;
  __syncthreads();
  if (tid < 16) {
    float m2 = -3.4e38f;
    for (int c2 = 0; c2 < 16; ++c2) m2 = fmaxf(m2, red[c2 * 16 + tid]);
    rowM[tid] = m2;
  }
  __syncthreads();
  const float rm = rowM[row];
  float sum = 0.f;
  for (int m = chk * 128; m < chk * 128 + 128; ++m) {
    const float e = __expf(sS[m * SROW + row] - rm);
    sS[m * SROW + row] = e;
    sum += e;
  }
  red[chk * 16 + row] = sum;
  __syncthreads();
  if (tid < 16) {
    float s2 = 0.f;
    for (int c2 = 0; c2 < 16; ++c2) s2 += red[c2 * 16 + tid];
    rowR[tid] = 1.0f / s2;
  }
  __syncthreads();

  // w[h,m] += sum_n attn[n,m]; per-m reads are 16 contiguous floats.
  float rrv[16];
#pragma unroll
  for (int r = 0; r < 16; ++r) rrv[r] = rowR[r];
  for (int m = tid; m < MM; m += 256) {
    const float* sp = &sS[m * SROW];
    float acc = 0.f;
#pragma unroll
    for (int r = 0; r < 16; ++r) acc += sp[r] * rrv[r];
    atomicAdd(&w[h * MM + m], acc);
  }
}

// ---------------- K4: pooled += (w_h · v) Wv[h] + (Σ w_h) bv[h] ----------------
__global__ __launch_bounds__(256) void k4_pool(
    const float* __restrict__ v, const float* __restrict__ Wv,
    const float* __restrict__ bvec, const float* __restrict__ w,
    float* __restrict__ pooled)
{
  __shared__ float part[4][64], wsum[4], sh[64], Wtot;
  const int tid = threadIdx.x, h = blockIdx.x;
  const int i = tid & 63, g = tid >> 6;
  float acc = 0.f, wp = 0.f;
  for (int m = g; m < MM; m += 4) {
    const float wm = w[h * MM + m];
    acc += wm * v[(size_t)m * 64 + i];
    wp += wm;
  }
  part[g][i] = acc;
  if (i == 0) wsum[g] = wp;
  __syncthreads();
  if (tid < 64) sh[tid] = part[0][tid] + part[1][tid] + part[2][tid] + part[3][tid];
  if (tid == 64) Wtot = wsum[0] + wsum[1] + wsum[2] + wsum[3];
  __syncthreads();
  for (int d = tid; d < Dm; d += 256) {
    float val = Wtot * bvec[h * Dm + d];
    for (int ii = 0; ii < 64; ++ii) val += sh[ii] * Wv[((size_t)h * 64 + ii) * Dm + d];
    atomicAdd(&pooled[d], val);
  }
}

// ---------------- K5: out = pooled @ Wo^T + bo ----------------
__global__ __launch_bounds__(256) void k5_out(
    const float* __restrict__ pooled, const float* __restrict__ Wo,
    const float* __restrict__ bo, float* __restrict__ out)
{
  __shared__ float pl[Dm];
  const int tid = threadIdx.x;
  for (int e = tid; e < Dm; e += 256) pl[e] = pooled[e];
  __syncthreads();
  for (int d = tid; d < Dm; d += 256) {
    float acc = bo[d];
    const float* wr = Wo + (size_t)d * Dm;
    for (int e = 0; e < Dm; ++e) acc += pl[e] * wr[e];
    out[d] = acc;
  }
}

static inline char* alignup(char* p, size_t a) {
  return (char*)(((uintptr_t)p + a - 1) & ~(uintptr_t)(a - 1));
}

extern "C" void kernel_launch(void* const* d_in, const int* in_sizes, int n_in,
                              void* d_out, int out_size, void* d_ws, size_t ws_size,
                              hipStream_t stream) {
  (void)in_sizes; (void)n_in; (void)out_size; (void)ws_size;
  const float* q   = (const float*)d_in[0];
  const float* k   = (const float*)d_in[1];
  const float* v   = (const float*)d_in[2];
  const float* Wq  = (const float*)d_in[3];
  const float* bq  = (const float*)d_in[4];
  const float* Wk  = (const float*)d_in[5];
  const float* bk  = (const float*)d_in[6];
  const float* Wv  = (const float*)d_in[7];
  const float* bv  = (const float*)d_in[8];
  const float* Wo  = (const float*)d_in[9];
  const float* bo  = (const float*)d_in[10];
  float* out = (float*)d_out;

  // Workspace carve (~5.2 MB total)
  char* p = alignup((char*)d_ws, 256);
  float*  G      = (float*)p;  p = alignup(p + (size_t)HH*64*64*4, 256);
  float*  av     = (float*)p;  p = alignup(p + (size_t)HH*64*4,    256);
  float*  bvw    = (float*)p;  p = alignup(p + (size_t)HH*64*4,    256);
  float*  cbv    = (float*)p;  p = alignup(p + (size_t)HH*4,       256);
  float*  u      = (float*)p;  p = alignup(p + (size_t)HH*NN*4,    256);
  float*  t      = (float*)p;  p = alignup(p + (size_t)HH*MM*4,    256);
  float*  w      = (float*)p;  p = alignup(p + (size_t)HH*MM*4,    256);
  float*  pooled = (float*)p;  p = alignup(p + (size_t)Dm*4,       256);
  __bf16* qG     = (__bf16*)p; p = alignup(p + (size_t)HH*NN*64*2, 256);
  __bf16* kbf    = (__bf16*)p; p = alignup(p + (size_t)MM*64*2,    256);

  k1_prep<<<HH, 256, 0, stream>>>(Wq, bq, Wk, bk, G, av, bvw, cbv, w, pooled);
  k2_qg<<<dim3(NN/64, HH), 256, 0, stream>>>(q, k, G, av, bvw, qG, kbf, u, t);
  const size_t k3_lds = (size_t)(MM * SROW + 256 + 16 + 16) * sizeof(float);
  k3_attn<<<dim3(NN/16, HH), 256, k3_lds, stream>>>(qG, kbf, u, t, cbv, w);
  k4_pool<<<HH, 256, 0, stream>>>(v, Wv, bv, w, pooled);
  k5_out<<<1, 256, 0, stream>>>(pooled, Wo, bo, out);
}
